// MultiHeadSelfAttention_27900107555206
// MI455X (gfx1250) — compile-verified, matrix operand order checked by reading, not run on hardware
//
#include <hip/hip_runtime.h>
#include <hip/hip_bf16.h>
#include <math.h>

// ---------------- types ----------------
typedef __attribute__((ext_vector_type(16))) _Float16 v16h;
typedef __attribute__((ext_vector_type(8)))  _Float16 v8h;
typedef __attribute__((ext_vector_type(8)))  float    v8f;
typedef unsigned int u32;
typedef __attribute__((ext_vector_type(4)))  u32 v4u;
typedef __attribute__((ext_vector_type(8)))  int v8i;
typedef __attribute__((ext_vector_type(4)))  int v4i;

#define BATCH   4
#define SEQ     2048
#define DIN     1024
#define NHEAD   16
#define HDIM    64
#define OTOT    3072          // 3 * 1024 (Wq|Wk|Wv concatenated)
#define XELEMS  ((size_t)BATCH * SEQ * DIN)        // 8388608
#define WELEMS  ((size_t)DIN * DIN)                // 1048576 per weight
#define HEADSZ  ((size_t)SEQ * HDIM)               // 131072 elems per (b,h)
#define QKVSZ   ((size_t)BATCH * NHEAD * HEADSZ)   // per matrix

// Load one 16-bit WMMA operand fragment (A-layout; B-layout is the mirror with
// lane = column) from a row-major LDS tile with `stride` halfs per row.
// ISA 16-bit layout: lanes 0-15 hold K={kb+0..7, kb+16..23}, lanes 16-31 hold
// K={kb+8..15, kb+24..31}, row/col = lane & 15.  => two 16B loads per lane.
__device__ inline v16h frag_ld(const _Float16* base, int stride, int lane) {
    const _Float16* p = base + (lane & 15) * stride + ((lane >> 4) << 3);
    v8h lo = *(const v8h*)(p);
    v8h hi = *(const v8h*)(p + 16);
    return __builtin_shufflevector(lo, hi, 0, 1, 2, 3, 4, 5, 6, 7,
                                   8, 9, 10, 11, 12, 13, 14, 15);
}

// ---------------- Tensor Data Mover: 2D tile -> LDS ----------------
// Builds a D# per CDNA5 ISA ch.8 and issues TENSOR_LOAD_TO_LDS.
//   g0: count=1 | lds_addr | global_addr[56:0] | type=2
//   g1: data_size=2B, pad_enable + pad codes, tensor/tile dims, dim0 stride
// padi: pad_interval code (3 => every 16 DW, 4 => every 32 DW)
// pada: pad_amount  code (3 => 4 DW = 8 halfs)
__device__ inline void tdm_load_2d(u32 lds_addr, const void* gptr,
                                   u32 padi, u32 pada,
                                   u32 tensor_d0, u32 tensor_d1,
                                   u32 tile_d0, u32 tile_d1, u32 stride0) {
    unsigned long long ga = (unsigned long long)gptr;
    v4u g0;
    g0[0] = 1u;                               // count=1, user mode, no gather
    g0[1] = lds_addr;                         // LDS byte address
    g0[2] = (u32)ga;                          // global addr [31:0]
    g0[3] = (u32)((ga >> 32) & 0x1FFFFFFu) | (2u << 30);  // addr[56:32] | type=2
    v8i g1;
    g1[0] = (int)((1u << 16) |                // data_size = 2 bytes
                  (1u << 20) |                // pad_enable
                  (padi << 22) | (pada << 25));
    g1[1] = (int)((tensor_d0 & 0xFFFFu) << 16);            // [79:48] tensor_dim0 (lo16)
    g1[2] = (int)((tensor_d0 >> 16) | ((tensor_d1 & 0xFFFFu) << 16));
    g1[3] = (int)((tensor_d1 >> 16) | (tile_d0 << 16));    // [127:112] tile_dim0
    g1[4] = (int)(tile_d1 & 0xFFFFu);                      // [143:128] tile_dim1, tile_dim2=0
    g1[5] = (int)stride0;                                  // tensor_dim0_stride lo32
    g1[6] = 0;                                             // stride hi, dim1_stride lo
    g1[7] = 0;
    v4i z4 = {0, 0, 0, 0};
#if __clang_major__ >= 23
    v8i z8 = {0, 0, 0, 0, 0, 0, 0, 0};
    __builtin_amdgcn_tensor_load_to_lds(g0, g1, z4, z4, z8, 0);
#else
    __builtin_amdgcn_tensor_load_to_lds(g0, g1, z4, z4, 0);
#endif
}

// ---------------- kernel 1: fp32 -> f16 convert ----------------
__global__ void cvt_f16(const float* __restrict__ x,
                        const float* __restrict__ wq,
                        const float* __restrict__ wk,
                        const float* __restrict__ wv,
                        _Float16* __restrict__ xh,
                        _Float16* __restrict__ wh) {
    size_t gid = (size_t)blockIdx.x * blockDim.x + threadIdx.x;
    size_t gsz = (size_t)gridDim.x * blockDim.x;
    for (size_t i = gid; i < XELEMS; i += gsz) xh[i] = (_Float16)x[i];
    for (size_t i = gid; i < WELEMS; i += gsz) {
        wh[i]              = (_Float16)wq[i];
        wh[WELEMS + i]     = (_Float16)wk[i];
        wh[2 * WELEMS + i] = (_Float16)wv[i];
    }
}

// ---------------- kernel 2: QKV projection GEMM ----------------
// C[m, o] = sum_k xh[m,k] * wh[o,k]   (NT GEMM, both row-major along K)
// Block tile 128(M) x 128(O); K staged 32 wide, DOUBLE-BUFFERED via the TDM
// (rows hardware-padded to 40 halfs); 8 waves, 32x64 output tile each.
// Q,K stored [mat][b][h][n][d]; V stored TRANSPOSED [b][h][d][n] so the
// attention kernel's PV operand is a plain row-major tile.
#define LDSTR 40   // padded row stride (halfs): 32 data + 8 pad
#define KTILES (DIN / 32)
__global__ __launch_bounds__(256) void qkv_gemm(const _Float16* __restrict__ xh,
                                                const _Float16* __restrict__ wh,
                                                _Float16* __restrict__ qkv) {
    __shared__ _Float16 Xs[2][128 * LDSTR];   // 2 x 10 KB
    __shared__ _Float16 Ws[2][128 * LDSTR];   // 2 x 10 KB

    const int tid  = threadIdx.x;
    const int lane = tid & 31;
    const int w    = tid >> 5;            // 0..7
    const int Mb   = blockIdx.x * 128;
    const int Ob   = blockIdx.y * 128;
    const int wm   = (w >> 1) * 32;       // wave row base within block tile
    const int wo   = (w & 1) * 64;        // wave col base within block tile

    const _Float16* xrow = xh + (size_t)Mb * DIN;
    const _Float16* wrow = wh + (size_t)Ob * DIN;
    u32 xs_lds[2] = { (u32)(unsigned long long)(void*)&Xs[0][0],
                      (u32)(unsigned long long)(void*)&Xs[1][0] };
    u32 ws_lds[2] = { (u32)(unsigned long long)(void*)&Ws[0][0],
                      (u32)(unsigned long long)(void*)&Ws[1][0] };

    // prologue: stage K-tile 0 into buffer 0 (waves 0/1 drive the TDM)
    if (w == 0) tdm_load_2d(xs_lds[0], xrow, 3u, 3u, DIN, BATCH * SEQ, 32u, 128u, DIN);
    if (w == 1) tdm_load_2d(ws_lds[0], wrow, 3u, 3u, DIN, OTOT,        32u, 128u, DIN);

    v8f acc[2][4];
    for (int i = 0; i < 2; ++i)
        for (int j = 0; j < 4; ++j)
            for (int e = 0; e < 8; ++e) acc[i][j][e] = 0.0f;

    for (int kt = 0; kt < KTILES; ++kt) {
        const int cur = kt & 1, nxt = cur ^ 1;
        if (kt + 1 < KTILES) {
            // prefetch next K-tile into the other buffer, then wait for the
            // OLDEST transfer only (current tile) -> DMA overlaps compute.
            if (w == 0) tdm_load_2d(xs_lds[nxt], xrow + (kt + 1) * 32,
                                    3u, 3u, DIN, BATCH * SEQ, 32u, 128u, DIN);
            if (w == 1) tdm_load_2d(ws_lds[nxt], wrow + (kt + 1) * 32,
                                    3u, 3u, DIN, OTOT, 32u, 128u, DIN);
            __builtin_amdgcn_s_wait_tensorcnt(1);
        } else {
            __builtin_amdgcn_s_wait_tensorcnt(0);
        }
        __syncthreads();   // publish: buffer `cur` is resident

        v16h a[2], b[4];
        a[0] = frag_ld(&Xs[cur][(wm +  0) * LDSTR], LDSTR, lane);
        a[1] = frag_ld(&Xs[cur][(wm + 16) * LDSTR], LDSTR, lane);
        for (int j = 0; j < 4; ++j)
            b[j] = frag_ld(&Ws[cur][(wo + j * 16) * LDSTR], LDSTR, lane);

        for (int i = 0; i < 2; ++i)
            for (int j = 0; j < 4; ++j)
                acc[i][j] = __builtin_amdgcn_wmma_f32_16x16x32_f16(
                    false, a[i], false, b[j], (short)0, acc[i][j], false, false);
        __syncthreads();   // all reads of `cur` done before it is re-filled
    }

    // epilogue: C layout -> lane (l&15)=col, vgpr r = row (+8 for lanes>=16)
    const int hs = lane >> 4, ln = lane & 15;
    for (int i = 0; i < 2; ++i)
        for (int j = 0; j < 4; ++j)
            for (int r = 0; r < 8; ++r) {
                int m = Mb + wm + i * 16 + r + 8 * hs;
                int o = Ob + wo + j * 16 + ln;
                int mat = o >> 10, oo = o & 1023;
                int h = oo >> 6, d = oo & 63;
                int bidx = m >> 11, n = m & 2047;
                size_t dst;
                if (mat == 2)   // V transposed: [b][h][d][n]
                    dst = 2 * QKVSZ + ((size_t)(bidx * NHEAD + h) * HDIM + d) * SEQ + n;
                else            // Q,K: [mat][b][h][n][d]
                    dst = (size_t)mat * QKVSZ +
                          ((size_t)(bidx * NHEAD + h)) * HEADSZ + (size_t)n * HDIM + d;
                qkv[dst] = (_Float16)acc[i][j][r];
            }
}

// ---------------- kernel 3: fused flash attention ----------------
// block = 128 threads (4 waves); grid = (N/64, NHEAD, BATCH).
// Each block: 64 query rows; loop over 64-wide K/V tiles with online softmax.
// K tile [n][d] and transposed V tile [d][n] staged DOUBLE-BUFFERED by the
// TDM with hardware row padding to 72 halfs.
#define KSTR 72    // padded row stride (halfs): 64 data + 8 pad
#define NTILES (SEQ / 64)
__global__ __launch_bounds__(128) void flash_attn(const _Float16* __restrict__ qkv,
                                                  float* __restrict__ out) {
    const int mt = blockIdx.x;   // 0..31 row tile
    const int h  = blockIdx.y;   // 0..15
    const int b  = blockIdx.z;   // 0..3

    __shared__ _Float16 Ks[2][64 * KSTR];    // K tile, row-major [n][d]
    __shared__ _Float16 Vt[2][64 * KSTR];    // V tile transposed [d][n]
    __shared__ _Float16 Ps[4][16 * KSTR];    // per-wave P tile [m][n]

    const int tid = threadIdx.x, lane = tid & 31, w = tid >> 5;
    const int ln = lane & 15, hs = lane >> 4;

    const _Float16* qh = qkv + ((size_t)(b * NHEAD + h)) * HEADSZ;
    const _Float16* kh = qh + QKVSZ;
    const _Float16* vt = qkv + 2 * QKVSZ + ((size_t)(b * NHEAD + h)) * HEADSZ; // [d][n]

    u32 ks_lds[2] = { (u32)(unsigned long long)(void*)&Ks[0][0],
                      (u32)(unsigned long long)(void*)&Ks[1][0] };
    u32 vt_lds[2] = { (u32)(unsigned long long)(void*)&Vt[0][0],
                      (u32)(unsigned long long)(void*)&Vt[1][0] };

    // prologue: stage K/V tile 0 into buffer 0 (overlaps the Q loads below)
    if (w == 0) tdm_load_2d(ks_lds[0], kh, 4u, 3u, HDIM, SEQ, 64u, 64u, HDIM);
    if (w == 1) tdm_load_2d(vt_lds[0], vt, 4u, 3u, SEQ, HDIM, 64u, 64u, SEQ);

    // Q fragments: wave w owns rows [w*16, w*16+16) of this 64-row tile.
    v16h qf[2];
    {
        const _Float16* qr = qh + (size_t)(mt * 64 + w * 16 + ln) * HDIM;
        for (int kd = 0; kd < 2; ++kd) {
            v8h lo = *(const v8h*)(qr + kd * 32 + hs * 8);
            v8h hi = *(const v8h*)(qr + kd * 32 + 16 + hs * 8);
            qf[kd] = __builtin_shufflevector(lo, hi, 0, 1, 2, 3, 4, 5, 6, 7,
                                             8, 9, 10, 11, 12, 13, 14, 15);
        }
    }

    float rowmax[8], rowsum[8];
    v8f o_acc[4];
    for (int r = 0; r < 8; ++r) { rowmax[r] = -3.0e38f; rowsum[r] = 0.0f; }
    for (int j = 0; j < 4; ++j)
        for (int e = 0; e < 8; ++e) o_acc[j][e] = 0.0f;

    for (int nt = 0; nt < NTILES; ++nt) {
        const int cur = nt & 1, nxt = cur ^ 1;
        if (nt + 1 < NTILES) {
            if (w == 0) tdm_load_2d(ks_lds[nxt], kh + (size_t)(nt + 1) * 64 * HDIM,
                                    4u, 3u, HDIM, SEQ, 64u, 64u, HDIM);
            if (w == 1) tdm_load_2d(vt_lds[nxt], vt + (size_t)(nt + 1) * 64,
                                    4u, 3u, SEQ, HDIM, 64u, 64u, SEQ);
            __builtin_amdgcn_s_wait_tensorcnt(1);
        } else {
            __builtin_amdgcn_s_wait_tensorcnt(0);
        }
        __syncthreads();   // buffer `cur` resident

        // S = Q * K^T  (16x64 per wave, k over head dim 64 = 2 WMMA steps)
        v8f s[4];
        for (int j = 0; j < 4; ++j)
            for (int e = 0; e < 8; ++e) s[j][e] = 0.0f;
        for (int kd = 0; kd < 2; ++kd)
            for (int j = 0; j < 4; ++j) {
                v16h bf = frag_ld(&Ks[cur][(j * 16) * KSTR + kd * 32], KSTR, lane);
                s[j] = __builtin_amdgcn_wmma_f32_16x16x32_f16(
                    false, qf[kd], false, bf, (short)0, s[j], false, false);
            }

        // online softmax: rows live at vgpr r (+8 for lanes>=16),
        // columns spread over the 16 lanes of each half x 4 accumulators.
        for (int r = 0; r < 8; ++r) {
            float sv[4], mx = -3.0e38f;
            for (int j = 0; j < 4; ++j) {
                sv[j] = s[j][r] * 0.125f;       // 1/sqrt(64)
                mx = fmaxf(mx, sv[j]);
            }
            for (int msk = 1; msk < 16; msk <<= 1)
                mx = fmaxf(mx, __shfl_xor(mx, msk, 32));
            float nm = fmaxf(rowmax[r], mx);
            float alpha = __expf(rowmax[r] - nm);
            rowmax[r] = nm;
            float ps = 0.0f;
            for (int j = 0; j < 4; ++j) {
                float p = __expf(sv[j] - nm);
                ps += p;
                Ps[w][(r + 8 * hs) * KSTR + j * 16 + ln] = (_Float16)p;
            }
            for (int msk = 1; msk < 16; msk <<= 1)
                ps += __shfl_xor(ps, msk, 32);
            rowsum[r] = rowsum[r] * alpha + ps;
            for (int j = 0; j < 4; ++j) o_acc[j][r] *= alpha;
        }

        // O += P * V   (A = P from wave-private LDS, B = V^T rows)
        for (int kn = 0; kn < 2; ++kn) {
            v16h pf = frag_ld(&Ps[w][kn * 32], KSTR, lane);
            for (int jd = 0; jd < 4; ++jd) {
                v16h vf = frag_ld(&Vt[cur][(jd * 16) * KSTR + kn * 32], KSTR, lane);
                o_acc[jd] = __builtin_amdgcn_wmma_f32_16x16x32_f16(
                    false, pf, false, vf, (short)0, o_acc[jd], false, false);
            }
        }
        __syncthreads();   // reads of `cur` done before it is re-filled
    }

    // epilogue: out[b, n, h*64 + d] fp32
    for (int jd = 0; jd < 4; ++jd)
        for (int r = 0; r < 8; ++r) {
            int n   = mt * 64 + w * 16 + r + 8 * hs;
            int col = h * HDIM + jd * 16 + ln;
            out[((size_t)(b * SEQ + n)) * DIN + col] = o_acc[jd][r] / rowsum[r];
        }
}

// ---------------- launch ----------------
extern "C" void kernel_launch(void* const* d_in, const int* in_sizes, int n_in,
                              void* d_out, int out_size, void* d_ws, size_t ws_size,
                              hipStream_t stream) {
    (void)in_sizes; (void)n_in; (void)out_size; (void)ws_size;
    const float* x  = (const float*)d_in[0];
    // d_in[1] = attn_mask: expanded but never applied in the reference -> unused
    const float* wq = (const float*)d_in[2];
    const float* wk = (const float*)d_in[3];
    const float* wv = (const float*)d_in[4];
    float* out = (float*)d_out;

    _Float16* xh   = (_Float16*)d_ws;                       // 16 MB
    _Float16* wh   = xh + XELEMS;                           //  6 MB
    _Float16* qkvh = wh + (size_t)OTOT * DIN;               // 48 MB

    cvt_f16<<<2048, 256, 0, stream>>>(x, wq, wk, wv, xh, wh);

    dim3 g1(BATCH * SEQ / 128, OTOT / 128);                 // 64 x 24
    qkv_gemm<<<g1, 256, 0, stream>>>(xh, wh, qkvh);

    dim3 g2(SEQ / 64, NHEAD, BATCH);                        // 32 x 16 x 4
    flash_attn<<<g2, 128, 0, stream>>>(qkvh, out);
}
